// MyGCN_33621003993517
// MI455X (gfx1250) — matrix operand order repaired and checked by reading, not compiled, not run on hardware
//
#include <hip/hip_runtime.h>
#include <hip/hip_bf16.h>

typedef __attribute__((ext_vector_type(16))) _Float16 v16h;
typedef __attribute__((ext_vector_type(8)))  _Float16 v8h;
typedef __attribute__((ext_vector_type(4)))  _Float16 v4h;
typedef __attribute__((ext_vector_type(8)))  float    v8f;
typedef __attribute__((ext_vector_type(4)))  float    v4f;

#define NN   8192   // nodes
#define INP  512
#define OUTD 256
#define BM   64     // rows per workgroup
#define BK   32     // K step (matches wmma f16 K); also y block size
#define LDA  40     // LDS stride (halves), padded
#define LDB  40
#define YBLK (OUTD * BK)        // halves per y block (16 KB)
#define ASH  (BM * LDA)         // halves per As buffer
#define BSH  (OUTD * LDB)       // halves per Bs buffer

// ---------------------------------------------------------------------------
// Kernel 1: degrees -> dinv[i] = rsqrt(1 + popcount(adj[i,:] > 0.5))
// ---------------------------------------------------------------------------
__global__ __launch_bounds__(256)
void gcn_deg_kernel(const float* __restrict__ adj, float* __restrict__ dinv) {
    __shared__ float red[256];
    const int row = blockIdx.x;
    const int t   = threadIdx.x;
    float cnt = 0.0f;
    const float* rp = adj + (size_t)row * NN;
    for (int c = t * 4; c < NN; c += 256 * 4) {
        v4f v = __builtin_nontemporal_load((const v4f*)(rp + c));
        cnt += (v[0] > 0.5f) ? 1.0f : 0.0f;
        cnt += (v[1] > 0.5f) ? 1.0f : 0.0f;
        cnt += (v[2] > 0.5f) ? 1.0f : 0.0f;
        cnt += (v[3] > 0.5f) ? 1.0f : 0.0f;
    }
    red[t] = cnt;
    __syncthreads();
    for (int s = 128; s > 0; s >>= 1) {
        if (t < s) red[t] += red[t + s];
        __syncthreads();
    }
    if (t == 0) dinv[row] = rsqrtf(red[0] + 1.0f);   // +1 self loop
}

// ---------------------------------------------------------------------------
// Kernel 2: yB = dinv ⊙ (x @ w) in K-blocked f16 layout:
//   element(row g, col n) -> yB[(g>>5)*YBLK + n*32 + (g&31)]
// so one 32-row K-tile is a contiguous 16 KB block (async-copy friendly).
// ---------------------------------------------------------------------------
__global__ __launch_bounds__(256)
void gcn_xw_kernel(const float* __restrict__ x, const float* __restrict__ w,
                   const float* __restrict__ dinv, _Float16* __restrict__ yB) {
    __shared__ _Float16 As[BM * LDA];
    __shared__ _Float16 Bs[OUTD * LDB];
    const int t = threadIdx.x;
    const int lane = t & 31, wave = t >> 5;
    const int n = lane & 15, hi = lane >> 4;
    const int wr = wave >> 1, wc = wave & 1;
    const int rowBase = blockIdx.x * BM;

    v8f acc[8] = {};
    const int ar = t >> 2, ac0 = (t & 3) * 8;    // A stage: 64 rows x 32 cols
    const int bg = t >> 6, bc0 = (t & 63) * 4;   // B stage: 8 k-rows x 4 n-cols

    for (int kk = 0; kk < INP; kk += BK) {
        // stage A: x tile (f32 -> f16)
        {
            const float* src = x + (size_t)(rowBase + ar) * INP + kk + ac0;
            v4f f0 = *(const v4f*)(src);
            v4f f1 = *(const v4f*)(src + 4);
            v8h o8;
            #pragma unroll
            for (int i = 0; i < 4; ++i) { o8[i] = (_Float16)f0[i]; o8[4 + i] = (_Float16)f1[i]; }
            *(v8h*)(&As[ar * LDA + ac0]) = o8;
        }
        // stage B via 8x4 register transpose: Bs[o][c] = w[kk+c][o]
        {
            v4h wv[8];
            #pragma unroll
            for (int rr = 0; rr < 8; ++rr) {
                v4f wf = *(const v4f*)(w + (size_t)(kk + bg * 8 + rr) * OUTD + bc0);
                #pragma unroll
                for (int j = 0; j < 4; ++j) wv[rr][j] = (_Float16)wf[j];
            }
            #pragma unroll
            for (int j = 0; j < 4; ++j) {
                v8h col;
                #pragma unroll
                for (int rr = 0; rr < 8; ++rr) col[rr] = wv[rr][j];
                *(v8h*)(&Bs[(bc0 + j) * LDB + bg * 8]) = col;
            }
        }
        __syncthreads();

        const int am = wr * 16 + n;
        v8h alo = *(const v8h*)(&As[am * LDA + hi * 8]);
        v8h ahi = *(const v8h*)(&As[am * LDA + 16 + hi * 8]);
        v16h af = __builtin_shufflevector(alo, ahi, 0,1,2,3,4,5,6,7,8,9,10,11,12,13,14,15);
        #pragma unroll
        for (int j = 0; j < 8; ++j) {
            const int bn = (wc * 8 + j) * 16 + n;
            v8h blo = *(const v8h*)(&Bs[bn * LDB + hi * 8]);
            v8h bhi = *(const v8h*)(&Bs[bn * LDB + 16 + hi * 8]);
            v16h bf = __builtin_shufflevector(blo, bhi, 0,1,2,3,4,5,6,7,8,9,10,11,12,13,14,15);
            acc[j] = __builtin_amdgcn_wmma_f32_16x16x32_f16(
                false, af, false, bf, (short)0, acc[j], false, false);
        }
        __syncthreads();
    }
    float dr[8];
    #pragma unroll
    for (int r = 0; r < 8; ++r) dr[r] = dinv[rowBase + wr * 16 + hi * 8 + r];
    const int g0 = rowBase + wr * 16 + hi * 8;   // 8 consecutive rows, same 32-block
    #pragma unroll
    for (int j = 0; j < 8; ++j) {
        const int gcol = (wc * 8 + j) * 16 + n;
        v8h o;
        #pragma unroll
        for (int r = 0; r < 8; ++r) o[r] = (_Float16)(acc[j][r] * dr[r]);
        _Float16* dst = yB + (size_t)(g0 >> 5) * YBLK + (size_t)gcol * BK + (g0 & 31);
        *(v8h*)dst = o;   // one b128 store per tile
    }
}

// ---------------------------------------------------------------------------
// Kernel 3: h = relu(dinv ⊙ (A_bin @ y))   (M=8192, K=8192, N=256)
// Fully double-buffered (As and Bs), ONE barrier per K step.
// adj: register-pipelined non-temporal stream (read once).
// y:   CDNA5 async copy global->LDS (ASYNCcnt), contiguous 16 KB per tile.
// ---------------------------------------------------------------------------
__global__ __launch_bounds__(256)
void gcn_mp_kernel(const float* __restrict__ adj, const _Float16* __restrict__ yB,
                   const float* __restrict__ dinv, _Float16* __restrict__ h) {
    __shared__ _Float16 As[2 * ASH];
    __shared__ _Float16 Bs[2 * BSH];
    const int t = threadIdx.x;
    const int lane = t & 31, wave = t >> 5;
    const int n = lane & 15, hi = lane >> 4;
    const int wr = wave >> 1, wc = wave & 1;
    const int rowBase = blockIdx.x * BM;

    v8f acc[8] = {};
    const int ar = t >> 2, ac0 = (t & 3) * 8;    // A stage: 64 rows x 32 cols

    const float* aRow = adj + (size_t)(rowBase + ar) * NN + ac0;
    const unsigned bsBase = (unsigned)(size_t)(&Bs[0]);   // LDS byte offset (low 32 bits)

    // issue async copy of y block kb into Bs buffer `buf` (4 x b128 per thread)
    auto issueB = [&](int kb, int buf) {
        #pragma unroll
        for (int i = 0; i < 4; ++i) {
            const int gi = t + 256 * i;                       // 16-byte chunk id, coalesced
            const _Float16* src = yB + (size_t)kb * YBLK + (size_t)gi * 8;
            const unsigned dst = bsBase + (unsigned)buf * (BSH * 2)
                               + (unsigned)((gi >> 2) * (LDB * 2) + (gi & 3) * 16);
            asm volatile("global_load_async_to_lds_b128 %0, %1, off"
                         :: "v"(dst), "v"(src) : "memory");
        }
    };

    // prologue: tile 0 in flight
    issueB(0, 0);
    v4f a0 = __builtin_nontemporal_load((const v4f*)(aRow));
    v4f a1 = __builtin_nontemporal_load((const v4f*)(aRow + 4));

    for (int kk = 0; kk < NN; kk += BK) {
        const int ib = (kk >> 5) & 1;
        // ---- adj regs -> As[ib] (binarize; uniform-branch diagonal) ----
        {
            float fv[8] = {a0[0], a0[1], a0[2], a0[3], a1[0], a1[1], a1[2], a1[3]};
            v8h o8;
            #pragma unroll
            for (int i = 0; i < 8; ++i)
                o8[i] = (_Float16)((fv[i] > 0.5f) ? 1.0f : 0.0f);
            if ((unsigned)(kk - rowBase) < (unsigned)BM) {   // uniform: tile touches diagonal
                #pragma unroll
                for (int i = 0; i < 8; ++i)
                    if (rowBase + ar == kk + ac0 + i) o8[i] += (_Float16)1.0f;  // + I
            }
            *(v8h*)(&As[ib * ASH + ar * LDA + ac0]) = o8;
        }
        // ---- prefetch next adj tile into registers ----
        const int kn = (kk + BK < NN) ? (kk + BK) : 0;
        a0 = __builtin_nontemporal_load((const v4f*)(aRow + kn));
        a1 = __builtin_nontemporal_load((const v4f*)(aRow + kn + 4));

        // B(kk) is the only outstanding async batch: retire it, then sync.
        asm volatile("s_wait_asynccnt 0x0" ::: "memory");
        __syncthreads();

        // ---- now Bs[ib^1] is reusable by everyone: start B(kk+BK) ----
        issueB(kn >> 5, ib ^ 1);

        // ---- compute ----
        const _Float16* as = &As[ib * ASH];
        const _Float16* bs = &Bs[ib * BSH];
        const int am = wr * 16 + n;
        v8h alo = *(const v8h*)(&as[am * LDA + hi * 8]);
        v8h ahi = *(const v8h*)(&as[am * LDA + 16 + hi * 8]);
        v16h af = __builtin_shufflevector(alo, ahi, 0,1,2,3,4,5,6,7,8,9,10,11,12,13,14,15);
        #pragma unroll
        for (int j = 0; j < 8; ++j) {
            const int bn = (wc * 8 + j) * 16 + n;
            v8h blo = *(const v8h*)(&bs[bn * LDB + hi * 8]);
            v8h bhi = *(const v8h*)(&bs[bn * LDB + 16 + hi * 8]);
            v16h bf = __builtin_shufflevector(blo, bhi, 0,1,2,3,4,5,6,7,8,9,10,11,12,13,14,15);
            acc[j] = __builtin_amdgcn_wmma_f32_16x16x32_f16(
                false, af, false, bf, (short)0, acc[j], false, false);
        }
        // no second barrier: next iter writes the OTHER As/Bs buffers.
    }
    float dr[8];
    #pragma unroll
    for (int r = 0; r < 8; ++r) dr[r] = dinv[rowBase + wr * 16 + hi * 8 + r];
    #pragma unroll
    for (int j = 0; j < 8; ++j) {
        const int gcol = (wc * 8 + j) * 16 + n;
        #pragma unroll
        for (int r = 0; r < 8; ++r) {
            const int grow = rowBase + wr * 16 + hi * 8 + r;
            float v = acc[j][r] * dr[r];
            v = v > 0.0f ? v : 0.0f;                       // relu
            h[(size_t)grow * OUTD + gcol] = (_Float16)v;
        }
    }
}

// ---------------------------------------------------------------------------
// Kernel 4: out = h @ lin_w^T + lin_b  (M=8192, K=256, N=256), f32 out
// ---------------------------------------------------------------------------
__global__ __launch_bounds__(256)
void gcn_out_kernel(const _Float16* __restrict__ h, const float* __restrict__ lin_w,
                    const float* __restrict__ lin_b, float* __restrict__ out) {
    __shared__ _Float16 As[BM * LDA];
    __shared__ _Float16 Bs[OUTD * LDB];
    const int t = threadIdx.x;
    const int lane = t & 31, wave = t >> 5;
    const int n = lane & 15, hi = lane >> 4;
    const int wr = wave >> 1, wc = wave & 1;
    const int rowBase = blockIdx.x * BM;

    v8f acc[8] = {};
    const int ar = t >> 2, ac0 = (t & 3) * 8;

    for (int kk = 0; kk < OUTD; kk += BK) {
        // stage A: h tile (already f16, straight copy)
        {
            v8h hv = *(const v8h*)(h + (size_t)(rowBase + ar) * OUTD + kk + ac0);
            *(v8h*)(&As[ar * LDA + ac0]) = hv;
        }
        // stage B: one lin_w row per thread, 32 contiguous cols, f32 -> f16
        {
            const float* src = lin_w + (size_t)t * OUTD + kk;
            #pragma unroll
            for (int q = 0; q < 4; ++q) {
                v4f w0 = *(const v4f*)(src + q * 8);
                v4f w1 = *(const v4f*)(src + q * 8 + 4);
                v8h o8;
                #pragma unroll
                for (int j = 0; j < 4; ++j) { o8[j] = (_Float16)w0[j]; o8[4 + j] = (_Float16)w1[j]; }
                *(v8h*)(&Bs[t * LDB + q * 8]) = o8;
            }
        }
        __syncthreads();

        const int am = wr * 16 + n;
        v8h alo = *(const v8h*)(&As[am * LDA + hi * 8]);
        v8h ahi = *(const v8h*)(&As[am * LDA + 16 + hi * 8]);
        v16h af = __builtin_shufflevector(alo, ahi, 0,1,2,3,4,5,6,7,8,9,10,11,12,13,14,15);
        #pragma unroll
        for (int j = 0; j < 8; ++j) {
            const int bn = (wc * 8 + j) * 16 + n;
            v8h blo = *(const v8h*)(&Bs[bn * LDB + hi * 8]);
            v8h bhi = *(const v8h*)(&Bs[bn * LDB + 16 + hi * 8]);
            v16h bf = __builtin_shufflevector(blo, bhi, 0,1,2,3,4,5,6,7,8,9,10,11,12,13,14,15);
            acc[j] = __builtin_amdgcn_wmma_f32_16x16x32_f16(
                false, af, false, bf, (short)0, acc[j], false, false);
        }
        __syncthreads();
    }
    #pragma unroll
    for (int j = 0; j < 8; ++j) {
        const int gcol = (wc * 8 + j) * 16 + n;
        const float bj = lin_b[gcol];
        #pragma unroll
        for (int r = 0; r < 8; ++r) {
            const int grow = rowBase + wr * 16 + hi * 8 + r;
            out[(size_t)grow * OUTD + gcol] = acc[j][r] + bj;
        }
    }
}

// ---------------------------------------------------------------------------
extern "C" void kernel_launch(void* const* d_in, const int* in_sizes, int n_in,
                              void* d_out, int out_size, void* d_ws, size_t ws_size,
                              hipStream_t stream) {
    const float* x     = (const float*)d_in[0];   // [8192, 512]
    const float* adj   = (const float*)d_in[1];   // [8192, 8192]
    const float* w     = (const float*)d_in[2];   // [512, 256]
    const float* lin_w = (const float*)d_in[3];   // [256, 256]
    const float* lin_b = (const float*)d_in[4];   // [256]
    float* out = (float*)d_out;                   // [8192, 256] f32

    // workspace: dinv (32KB) | yB f16 blocked (4MB) | h f16 (4MB)  ~= 8.4 MB
    char* ws = (char*)d_ws;
    float*    dinv = (float*)ws;
    _Float16* yB   = (_Float16*)(ws + 32768);
    _Float16* hbuf = (_Float16*)(ws + 32768 + (size_t)NN * OUTD * 2);

    gcn_deg_kernel<<<NN, 256, 0, stream>>>(adj, dinv);
    gcn_xw_kernel <<<NN / BM, 256, 0, stream>>>(x, w, dinv, yB);
    gcn_mp_kernel <<<NN / BM, 256, 0, stream>>>(adj, yB, dinv, hbuf);
    gcn_out_kernel<<<NN / BM, 256, 0, stream>>>(hbuf, lin_w, lin_b, out);
}
